// Dist_Weighted_UDF_10376640987893
// MI455X (gfx1250) — matrix operand order repaired and checked by reading, not compiled, not run on hardware
//
#include <hip/hip_runtime.h>
#include <math.h>

typedef _Float16 f16x8  __attribute__((ext_vector_type(8)));
typedef _Float16 f16x16 __attribute__((ext_vector_type(16)));
typedef float    f32x8  __attribute__((ext_vector_type(8)));

// ---------------- problem constants ----------------
constexpr int NB   = 2;
constexpr int NPTS = 8192;
constexpr int NQ   = 8192;
constexpr int KK   = 10;     // KNN neighbors
constexpr int QT   = 4;      // queries per MLP workgroup
constexpr int ROWS = 48;     // QT*KK = 40, padded to 3 x 16-row tiles

// ---------------- layer tables (9 logical GEMMs) ----------------
// l0..l3 : patch MLP (6->64->128->128->128)
// l4     : attn L1, AB part (cv1+kdist+feat, padded 160ch -> 256)
// l5     : attn L1, C part  (patch_rep 128 -> 256, raw, no coeff)
// l6..l8 : attn 256->128->32->1
constexpr int L_KC[9]   = {1,2,4,4,5,4,8,4,1};           // k-chunks of 32
constexpr int L_NT[9]   = {4,8,8,8,16,16,8,2,1};         // n-tiles of 16
constexpr int L_CIN[9]  = {6,64,128,128,263,263,256,128,32}; // real W row length
constexpr int L_COUT[9] = {64,128,128,128,256,256,128,32,1};
constexpr int L_TYPE[9] = {0,0,0,0,1,2,0,0,0};           // cin remap kind
constexpr int L_BN[9]   = {1,1,1,1,1,0,1,1,0};
constexpr int L_WOFF[10]= {0,2048,10240,26624,43008,83968,116736,149504,153600,154112};
constexpr int L_COFF[10]= {0,64,192,320,448,704,960,1088,1120,1136};
constexpr int WPACK_HALVES = 154112;
constexpr int COEFF_PAIRS  = 1136;
constexpr size_t WPACK_BYTES = (size_t)WPACK_HALVES * 2;
constexpr size_t COEFF_BYTES = (size_t)COEFF_PAIRS * 2 * 4;

struct PackParams {
  const float* w[9];
  const float* b[9];
  const float* g[9];
  const float* be[9];
  const float* m[9];
  const float* v[9];
};

// ============ kernel 1: pack weights to WMMA-B layout + fold BN ============
// B-frag layout (16x16x32 f16): block = kc*ntiles+nt (512 f16), lane owns
// column (lane&15), 16 contiguous K values: K = kc*32 + (lane>=16?16:0) + e.
__global__ __launch_bounds__(256) void pack_kernel(PackParams pp,
                                                   _Float16* __restrict__ wpack,
                                                   float* __restrict__ coeff) {
  int gid = blockIdx.x * blockDim.x + threadIdx.x;
  if (gid < WPACK_HALVES) {
    int l = 0;
    #pragma unroll
    for (int t = 1; t < 9; ++t) if (gid >= L_WOFF[t]) l = t;
    int r      = gid - L_WOFF[l];
    int block  = r >> 9;
    int within = r & 511;
    int lane   = within >> 4;
    int el     = within & 15;
    int nt     = block % L_NT[l];
    int kc     = block / L_NT[l];
    int cout   = nt * 16 + (lane & 15);
    int pc     = kc * 32 + ((lane >> 4) << 4) + el;   // padded input channel
    int cin;
    if (L_TYPE[l] == 0)      cin = pc;
    else if (L_TYPE[l] == 1) cin = (pc < 32) ? (pc < 7 ? pc : -1) : (7 + pc - 32);
    else                     cin = 135 + pc;          // patch_rep part of cv2
    float val = 0.f;
    if (cin >= 0 && cin < L_CIN[l] && cout < L_COUT[l])
      val = pp.w[l][cout * L_CIN[l] + cin];
    wpack[gid] = (_Float16)val;
  }
  if (gid < COEFF_PAIRS) {
    int l = 0;
    #pragma unroll
    for (int t = 1; t < 9; ++t) if (gid >= L_COFF[t]) l = t;
    int c = gid - L_COFF[l];
    float alpha = 0.f, beta = 0.f;
    if (l != 5 && c < L_COUT[l]) {
      float bb = pp.b[l][c];
      if (L_BN[l]) {
        float s = pp.g[l][c] * rsqrtf(pp.v[l][c] + 1e-5f);
        alpha = s;
        beta  = (bb - pp.m[l][c]) * s + pp.be[l][c];
      } else { alpha = 1.f; beta = bb; }
    }
    coeff[2 * gid]     = alpha;
    coeff[2 * gid + 1] = beta;
  }
}

// ============ kernel 2: brute-force KNN (one wave32 per query) ============
__global__ __launch_bounds__(256) void knn_kernel(const float* __restrict__ pcd,
                                                  const float* __restrict__ qry,
                                                  float* __restrict__ cvws) {
  const int lane = threadIdx.x & 31;
  const int wq   = blockIdx.x * 8 + (threadIdx.x >> 5);  // global query id
  const int b    = wq >> 13;                             // / 8192
  const float* pc = pcd + (size_t)b * NPTS * 3;
  const float qx = qry[wq * 3 + 0];
  const float qy = qry[wq * 3 + 1];
  const float qz = qry[wq * 3 + 2];

  float bd[10];
  int   bi[10];
  #pragma unroll
  for (int j = 0; j < 10; ++j) { bd[j] = 3.0e38f; bi[j] = 0; }

  // per-lane sorted top-10 over strided points (branchless insert chain)
  for (int base = 0; base < NPTS; base += 32) {
    int i = base + lane;
    float dx = qx - pc[i * 3 + 0];
    float dy = qy - pc[i * 3 + 1];
    float dz = qz - pc[i * 3 + 2];
    float cd = dx * dx + dy * dy + dz * dz;
    int   ci = i;
    #pragma unroll
    for (int j = 0; j < 10; ++j) {
      bool sw = cd < bd[j];
      float td = bd[j]; int ti = bi[j];
      bd[j] = sw ? cd : bd[j]; bi[j] = sw ? ci : bi[j];
      cd    = sw ? td : cd;    ci    = sw ? ti : ci;
    }
  }

  // 10 rounds: pop global minimum across the wave (lexicographic tie-break)
  int myIdx = 0;
  for (int r = 0; r < 10; ++r) {
    float d = bd[0]; int idx = bi[0]; int src = lane;
    #pragma unroll
    for (int off = 16; off >= 1; off >>= 1) {
      float od = __shfl_xor(d, off);
      int   oi = __shfl_xor(idx, off);
      int   os = __shfl_xor(src, off);
      bool take = (od < d) || (od == d && os < src);
      d = take ? od : d; idx = take ? oi : idx; src = take ? os : src;
    }
    if (lane == r) myIdx = idx;
    if (lane == src) {                 // winner pops its head
      #pragma unroll
      for (int j = 0; j < 9; ++j) { bd[j] = bd[j + 1]; bi[j] = bi[j + 1]; }
      bd[9] = 3.0e38f; bi[9] = 0;
    }
  }

  if (lane < KK) {
    float px = pc[myIdx * 3 + 0], py = pc[myIdx * 3 + 1], pz = pc[myIdx * 3 + 2];
    float lx = qx - px, ly = qy - py, lz = qz - pz;
    float kd = fabsf(lx) + fabsf(ly) + fabsf(lz);
    float* row = cvws + (size_t)(wq * KK + lane) * 8;
    row[0] = lx; row[1] = ly; row[2] = lz;
    row[3] = qx; row[4] = qy; row[5] = qz;
    row[6] = kd; row[7] = 0.f;
  }
}

// ============ WMMA fragment helpers ============
// A-frag (16-bit A, 16x32): lane row = lane&15; K = kc*32 + (lane>=16?16:0)
// giving one contiguous run [base..base+15] (VGPR0-7), matching the B pack.
__device__ __forceinline__ f16x16 load_a(const _Float16* buf, int stride,
                                         int mt, int kc, int lane) {
  int row = mt * 16 + (lane & 15);
  int kb  = kc * 32 + (lane & 16);
  const _Float16* p = buf + row * stride + kb;
  f16x8 lo = *(const f16x8*)(p);
  f16x8 hi = *(const f16x8*)(p + 8);
  f16x16 a;
  #pragma unroll
  for (int i = 0; i < 8; ++i) { a[i] = lo[i]; a[i + 8] = hi[i]; }
  return a;
}

__device__ __forceinline__ f16x16 load_b(const _Float16* wl, int block, int lane) {
  return *(const f16x16*)(wl + (block << 9) + (lane << 4));
}

// Generic tiled GEMM layer: LDS->LDS, fused alpha/beta (+LeakyReLU 0.2)
template <int MTILES, int NTILES, int KCHUNKS, bool LRELU>
__device__ __forceinline__ void run_layer(const _Float16* __restrict__ wl,
                                          const float* __restrict__ cl,
                                          const _Float16* __restrict__ inBuf, int inStride,
                                          _Float16* __restrict__ outBuf, int outStride,
                                          int wave, int lane) {
  const int col = lane & 15;
  const int rb0 = (lane & 16) >> 1;
  for (int t = wave; t < MTILES * NTILES; t += 8) {
    int mt = t / NTILES, nt = t % NTILES;
    f32x8 c = {};
    #pragma unroll
    for (int kc = 0; kc < KCHUNKS; ++kc) {
      f16x16 a = load_a(inBuf, inStride, mt, kc, lane);
      f16x16 b = load_b(wl, kc * NTILES + nt, lane);
      c = __builtin_amdgcn_wmma_f32_16x16x32_f16(false, a, false, b,
                                                 (short)0, c, false, false);
    }
    int ch = nt * 16 + col;
    float alpha = cl[2 * ch], beta = cl[2 * ch + 1];
    int rbase = mt * 16 + rb0;
    #pragma unroll
    for (int v2 = 0; v2 < 8; ++v2) {
      float y = alpha * c[v2] + beta;
      if (LRELU) y = (y >= 0.f) ? y : 0.2f * y;
      outBuf[(rbase + v2) * outStride + ch] = (_Float16)y;
    }
  }
}

// ============ kernel 3: fused dual-MLP + pool + attention + softmax ============
__global__ __launch_bounds__(256) void mlp_kernel(const _Float16* __restrict__ wpack,
                                                  const float* __restrict__ coeff,
                                                  const float* __restrict__ cvws,
                                                  float* __restrict__ out) {
  __shared__ _Float16 sIn[ROWS * 32];    // cv1+kdist (padded 32ch); later attn3 out
  __shared__ _Float16 sE [ROWS * 128];
  __shared__ _Float16 sF [ROWS * 128];
  __shared__ _Float16 sG [ROWS * 256];
  __shared__ _Float16 sP [16 * 128];     // patchfeat as one 16-row M-tile
  __shared__ float    sPV[5 * 256];      // per-query patch contribution (f32)
  __shared__ float    sKd[ROWS];
  __shared__ float    sScore[ROWS];

  const int tid  = threadIdx.x;
  const int lane = tid & 31;
  const int wave = tid >> 5;
  const int qbase   = blockIdx.x * QT;
  const int rowbase = qbase * KK;

  // ---- stage-in cv1 (pad to 32ch, zero-pad rows 40..47) + kdist ----
  for (int i = tid; i < ROWS * 32; i += 256) {
    int r = i >> 5, c = i & 31;
    float v = 0.f;
    if (r < QT * KK && c < 7) v = cvws[(size_t)(rowbase + r) * 8 + c];
    sIn[i] = (_Float16)v;
  }
  if (tid < ROWS) sKd[tid] = (tid < QT * KK) ? cvws[(size_t)(rowbase + tid) * 8 + 6] : 0.f;
  __syncthreads();

  // ---- patch MLP ----
  run_layer<3, 4, 1, true >(wpack + L_WOFF[0], coeff + 2 * L_COFF[0], sIn, 32,  sE, 128, wave, lane);
  __syncthreads();
  run_layer<3, 8, 2, true >(wpack + L_WOFF[1], coeff + 2 * L_COFF[1], sE, 128, sF, 128, wave, lane);
  __syncthreads();
  run_layer<3, 8, 4, true >(wpack + L_WOFF[2], coeff + 2 * L_COFF[2], sF, 128, sE, 128, wave, lane);
  __syncthreads();
  run_layer<3, 8, 4, false>(wpack + L_WOFF[3], coeff + 2 * L_COFF[3], sE, 128, sF, 128, wave, lane);
  __syncthreads();

  // ---- max-pool over K into sP (rows 0..QT-1), zero pad rows QT..15 ----
  for (int i = tid; i < 16 * 128; i += 256) {
    int r = i >> 7, c = i & 127;
    float mval = 0.f;
    if (r < QT) {
      mval = -3.0e38f;
      for (int k = 0; k < KK; ++k)
        mval = fmaxf(mval, (float)sF[(r * KK + k) * 128 + c]);
    }
    sP[i] = (_Float16)mval;
  }
  __syncthreads();

  // ---- PV = W_c @ patchfeat (raw f32, per-query) ----
  {
    const _Float16* wl = wpack + L_WOFF[5];
    int col = lane & 15, rb0 = (lane & 16) >> 1;
    for (int nt = wave; nt < 16; nt += 8) {
      f32x8 c = {};
      #pragma unroll
      for (int kc = 0; kc < 4; ++kc) {
        f16x16 a = load_a(sP, 128, 0, kc, lane);
        f16x16 b = load_b(wl, kc * 16 + nt, lane);
        c = __builtin_amdgcn_wmma_f32_16x16x32_f16(false, a, false, b,
                                                   (short)0, c, false, false);
      }
      #pragma unroll
      for (int v2 = 0; v2 < 8; ++v2) {
        int row = rb0 + v2;
        if (row < 5) sPV[row * 256 + nt * 16 + col] = c[v2];
      }
    }
  }
  __syncthreads();

  // ---- attn L1: [cv1+kdist(32) ; feat(128)] -> 256, + PV, coeff, LReLU ----
  {
    const _Float16* wl = wpack + L_WOFF[4];
    const float*    cl = coeff + 2 * L_COFF[4];
    int col = lane & 15, rb0 = (lane & 16) >> 1;
    for (int t = wave; t < 3 * 16; t += 8) {
      int mt = t >> 4, nt = t & 15;
      f32x8 c = {};
      {
        f16x16 a = load_a(sIn, 32, mt, 0, lane);
        f16x16 b = load_b(wl, 0 * 16 + nt, lane);
        c = __builtin_amdgcn_wmma_f32_16x16x32_f16(false, a, false, b,
                                                   (short)0, c, false, false);
      }
      #pragma unroll
      for (int kc = 1; kc < 5; ++kc) {
        f16x16 a = load_a(sF, 128, mt, kc - 1, lane);
        f16x16 b = load_b(wl, kc * 16 + nt, lane);
        c = __builtin_amdgcn_wmma_f32_16x16x32_f16(false, a, false, b,
                                                   (short)0, c, false, false);
      }
      int ch = nt * 16 + col;
      float alpha = cl[2 * ch], beta = cl[2 * ch + 1];
      int rbase = mt * 16 + rb0;
      #pragma unroll
      for (int v2 = 0; v2 < 8; ++v2) {
        int row = rbase + v2;
        float mm = c[v2] + sPV[(row / 10) * 256 + ch];
        float y  = alpha * mm + beta;
        y = (y >= 0.f) ? y : 0.2f * y;
        sG[row * 256 + ch] = (_Float16)y;
      }
    }
  }
  __syncthreads();

  run_layer<3, 8, 8, true>(wpack + L_WOFF[6], coeff + 2 * L_COFF[6], sG, 256, sE, 128, wave, lane);
  __syncthreads();
  run_layer<3, 2, 4, true>(wpack + L_WOFF[7], coeff + 2 * L_COFF[7], sE, 128, sIn, 32, wave, lane);
  __syncthreads();

  // ---- attn L4: 32 -> 1 (bias only), scores per row ----
  {
    const _Float16* wl = wpack + L_WOFF[8];
    const float beta = (coeff + 2 * L_COFF[8])[1];  // alpha == 1
    int col = lane & 15, rb0 = (lane & 16) >> 1;
    for (int mt = wave; mt < 3; mt += 8) {
      f16x16 a = load_a(sIn, 32, mt, 0, lane);
      f16x16 b = load_b(wl, 0, lane);
      f32x8 c = {};
      c = __builtin_amdgcn_wmma_f32_16x16x32_f16(false, a, false, b,
                                                 (short)0, c, false, false);
      if (col == 0) {
        #pragma unroll
        for (int v2 = 0; v2 < 8; ++v2)
          sScore[mt * 16 + rb0 + v2] = c[v2] + beta;
      }
    }
  }
  __syncthreads();

  // ---- softmax over K + weighted kdist sum ----
  if (tid < QT) {
    int base = tid * KK;
    float mval = -3.0e38f;
    for (int k = 0; k < KK; ++k) mval = fmaxf(mval, sScore[base + k]);
    float s = 0.f, dot = 0.f;
    for (int k = 0; k < KK; ++k) {
      float e = __expf(sScore[base + k] - mval);
      s += e;
      dot += e * sKd[base + k];
    }
    out[qbase + tid] = dot / s;
  }
}

// ============ launch ============
extern "C" void kernel_launch(void* const* d_in, const int* in_sizes, int n_in,
                              void* d_out, int out_size, void* d_ws, size_t ws_size,
                              hipStream_t stream) {
  const float* input_pcd = (const float*)d_in[0];
  const float* query     = (const float*)d_in[1];

  // flatten order: input_pcd, query_points, then per-layer dicts in insertion
  // order (w, b, [g, be, m, v]) for patch[4] then attn[4].
  PackParams pp = {};
  pp.w[0] = (const float*)d_in[2];  pp.b[0] = (const float*)d_in[3];
  pp.g[0] = (const float*)d_in[4];  pp.be[0] = (const float*)d_in[5];
  pp.m[0] = (const float*)d_in[6];  pp.v[0] = (const float*)d_in[7];
  pp.w[1] = (const float*)d_in[8];  pp.b[1] = (const float*)d_in[9];
  pp.g[1] = (const float*)d_in[10]; pp.be[1] = (const float*)d_in[11];
  pp.m[1] = (const float*)d_in[12]; pp.v[1] = (const float*)d_in[13];
  pp.w[2] = (const float*)d_in[14]; pp.b[2] = (const float*)d_in[15];
  pp.g[2] = (const float*)d_in[16]; pp.be[2] = (const float*)d_in[17];
  pp.m[2] = (const float*)d_in[18]; pp.v[2] = (const float*)d_in[19];
  pp.w[3] = (const float*)d_in[20]; pp.b[3] = (const float*)d_in[21];
  pp.w[4] = (const float*)d_in[22]; pp.b[4] = (const float*)d_in[23];
  pp.g[4] = (const float*)d_in[24]; pp.be[4] = (const float*)d_in[25];
  pp.m[4] = (const float*)d_in[26]; pp.v[4] = (const float*)d_in[27];
  pp.w[5] = (const float*)d_in[22];                       // shares attn L1 W
  pp.w[6] = (const float*)d_in[28]; pp.b[6] = (const float*)d_in[29];
  pp.g[6] = (const float*)d_in[30]; pp.be[6] = (const float*)d_in[31];
  pp.m[6] = (const float*)d_in[32]; pp.v[6] = (const float*)d_in[33];
  pp.w[7] = (const float*)d_in[34]; pp.b[7] = (const float*)d_in[35];
  pp.g[7] = (const float*)d_in[36]; pp.be[7] = (const float*)d_in[37];
  pp.m[7] = (const float*)d_in[38]; pp.v[7] = (const float*)d_in[39];
  pp.w[8] = (const float*)d_in[40]; pp.b[8] = (const float*)d_in[41];

  _Float16* wpack = (_Float16*)d_ws;
  float* coeff = (float*)((char*)d_ws + WPACK_BYTES);
  float* cvws  = (float*)((char*)d_ws + WPACK_BYTES + COEFF_BYTES);

  pack_kernel<<<(WPACK_HALVES + 255) / 256, 256, 0, stream>>>(pp, wpack, coeff);
  knn_kernel<<<(NB * NQ) / 8, 256, 0, stream>>>(input_pcd, query, cvws);
  mlp_kernel<<<(NB * NQ) / QT, 256, 0, stream>>>(wpack, coeff, cvws, (float*)d_out);
}
